// Mamba3Block_83846351552670
// MI455X (gfx1250) — matrix operand order repaired and checked by reading, not compiled, not run on hardware
//
#include <hip/hip_runtime.h>
#include <stdint.h>

// ---------------- Types / helpers ----------------
typedef __attribute__((ext_vector_type(16))) __bf16 bf16x16;
typedef __attribute__((ext_vector_type(8)))  float  f32x8;
typedef int v4i __attribute__((vector_size(16)));

union FragU { bf16x16 v; unsigned u[8]; };

__device__ __forceinline__ unsigned short f2bf(float f) {
  unsigned u = __builtin_bit_cast(unsigned, f);
  u += 0x7FFFu + ((u >> 16) & 1u);
  return (unsigned short)(u >> 16);
}
__device__ __forceinline__ float bf2f(unsigned short s) {
  return __builtin_bit_cast(float, (unsigned)s << 16);
}

// Dimensions
#define T_TOK   8192      // B*L
#define DM      768
#define DI      3072
#define NH      48
#define DH      64
#define PROJN   6192      // 2*DI + NH
#define R3      256
#define R2      1024
#define NE      8
#define CAP     8192      // max entries per expert
#define LSEQ    4096
#define NCH     64        // chunks per sequence
#define LCH     64        // chunk length

#define AS1 __attribute__((address_space(1)))
#define AS3 __attribute__((address_space(3)))

// ---------------- gfx1250 async global->LDS copy ----------------
#if __has_builtin(__builtin_amdgcn_global_load_async_to_lds_b128)
#define ASYNC_MODE 1
#else
#define ASYNC_MODE 2
#endif

__device__ __forceinline__ void async_copy16(const unsigned short* g,
                                             unsigned short* l) {
#if ASYNC_MODE == 1
  __builtin_amdgcn_global_load_async_to_lds_b128(
      (AS1 v4i*)(uintptr_t)g, (AS3 v4i*)l, 0, 0);
#else
  // vdst = LDS byte offset (low 32 bits of flat shared pointer), vaddr = 64b global
  asm volatile("global_load_async_to_lds_b128 %0, %1, off"
               :
               : "v"((unsigned)(uintptr_t)l), "v"((unsigned long long)(uintptr_t)g)
               : "memory");
#endif
}

template <int N>
__device__ __forceinline__ void async_wait() {
#if __has_builtin(__builtin_amdgcn_s_wait_asynccnt)
  __builtin_amdgcn_s_wait_asynccnt(N);
#else
  asm volatile("s_wait_asynccnt %0" ::"i"(N) : "memory");
#endif
}

// ---------------- WMMA 64x64 tile core ----------------
// lds A: [64][32] bf16 row-major (m,k). lds B: [64][32] bf16 (n,k).
__device__ __forceinline__ void wmma_step(const unsigned short* la,
                                          const unsigned short* lb,
                                          int m0, int lane, f32x8* acc) {
  const int nl = lane & 15;
  const int hf = lane >> 4;
  FragU a;
#pragma unroll
  for (int v = 0; v < 8; ++v) {
    // 16-bit A 16x32 layout: K = (v/4)*16 + half*8 + (v%4)*2 (+0,+1)
    int k0 = ((v >> 2) << 4) + (hf << 3) + ((v & 3) << 1);
    a.u[v] = *(const unsigned*)(la + (m0 + nl) * 32 + k0);
  }
#pragma unroll
  for (int nt = 0; nt < 4; ++nt) {
    const int n = nt * 16 + nl;
    FragU b;
#pragma unroll
    for (int v = 0; v < 8; ++v) {
      // 16-bit B 32x16 layout: K = (v/4)*16 + half*8 + (v%4)*2 (+0,+1), col = lane%16
      int k0 = ((v >> 2) << 4) + (hf << 3) + ((v & 3) << 1);
      b.u[v] = *(const unsigned*)(lb + n * 32 + k0);
    }
    acc[nt] = __builtin_amdgcn_wmma_f32_16x16x32_bf16(
        false, a.v, false, b.v, (short)0, acc[nt], false, false);
  }
}

// ---------------- Tile fills (async; 2 issues per thread per tile) ----------------
__device__ __forceinline__ void fill_direct(unsigned short* lds,
                                            const unsigned short* src, int ldK,
                                            size_t row0, int kbase, int tid) {
#pragma unroll
  for (int p = 0; p < 2; ++p) {
    int chunk = p * 128 + tid;
    int r = chunk >> 2, ko = (chunk & 3) << 3;
    async_copy16(src + (row0 + r) * (size_t)ldK + kbase + ko, lds + r * 32 + ko);
  }
}

// Pad rows are clamped (junk data, discarded by the guarded epilogue) so EXEC
// stays full and per-wave ASYNCcnt accounting is exact.
__device__ __forceinline__ void fill_clamp(unsigned short* lds,
                                           const unsigned short* src, int ldK,
                                           int row0, int nrows, int kbase, int tid) {
#pragma unroll
  for (int p = 0; p < 2; ++p) {
    int chunk = p * 128 + tid;
    int r = chunk >> 2, ko = (chunk & 3) << 3;
    int row = row0 + r;
    if (row >= nrows) row = nrows - 1;
    async_copy16(src + (size_t)row * ldK + kbase + ko, lds + r * 32 + ko);
  }
}

__device__ __forceinline__ void fill_gather(unsigned short* lds,
                                            const unsigned short* Lat,
                                            const int* tok, int mbase, int cnt,
                                            int kbase, int tid) {
#pragma unroll
  for (int p = 0; p < 2; ++p) {
    int chunk = p * 128 + tid;
    int r = chunk >> 2, ko = (chunk & 3) << 3;
    int entry = mbase + r;
    if (entry >= cnt) entry = cnt - 1;        // clamp: padded rows discarded later
    int t2 = tok[entry];
    async_copy16(Lat + (size_t)(t2 >> 1) * R3 + kbase + ko, lds + r * 32 + ko);
  }
}

// f32 (two top-k slots summed) -> bf16, through VGPRs (compute conversion).
__device__ __forceinline__ void fill_conv(unsigned short* lds, const float* Moe,
                                          size_t mbase, int kbase, int tid) {
#pragma unroll
  for (int p = 0; p < 2; ++p) {
    int chunk = p * 128 + tid;
    int r = chunk >> 2, ko = (chunk & 3) << 3;
    size_t row = mbase + r;
    const float* m0p = Moe + (row * 2) * (size_t)R2 + kbase + ko;
    const float* m1p = m0p + R2;
#pragma unroll
    for (int j = 0; j < 8; ++j) lds[r * 32 + ko + j] = f2bf(m0p[j] + m1p[j]);
  }
}

// ---------------- Weight transpose + bf16 convert ----------------
__global__ void k_transpose_bf16(const float* __restrict__ src,
                                 unsigned short* __restrict__ dst,
                                 int K, int N) {
  size_t zoff = (size_t)blockIdx.z * K * N;
  int n = blockIdx.x * 16 + threadIdx.x;
  int k = blockIdx.y * 16 + threadIdx.y;
  if (n < N && k < K)
    dst[zoff + (size_t)n * K + k] = f2bf(src[zoff + (size_t)k * N + n]);
}

// ---------------- RMSNorm 1 ----------------
__global__ __launch_bounds__(256) void k_rmsnorm(const float* __restrict__ x,
                                                 const float* __restrict__ w,
                                                 unsigned short* __restrict__ out) {
  __shared__ float red[256];
  int t = blockIdx.x, tid = threadIdx.x;
  const float* row = x + (size_t)t * DM;
  float v0 = row[tid], v1 = row[tid + 256], v2 = row[tid + 512];
  red[tid] = v0 * v0 + v1 * v1 + v2 * v2;
  __syncthreads();
  for (int s = 128; s > 0; s >>= 1) {
    if (tid < s) red[tid] += red[tid + s];
    __syncthreads();
  }
  float inv = rsqrtf(red[0] / (float)DM + 1e-5f);
  unsigned short* o = out + (size_t)t * DM;
  o[tid]       = f2bf(v0 * inv * w[tid]);
  o[tid + 256] = f2bf(v1 * inv * w[tid + 256]);
  o[tid + 512] = f2bf(v2 * inv * w[tid + 512]);
}

// ---------------- GEMM: proj = xn @ W_in, epilogue split u/z/dt ----------------
__global__ __launch_bounds__(128) void k_gemm_proj(
    const unsigned short* __restrict__ A, const unsigned short* __restrict__ Bt,
    const float* __restrict__ dt_bias, unsigned short* __restrict__ Ubf,
    unsigned short* __restrict__ Zsbf, float* __restrict__ Alpha) {
  __shared__ __align__(16) unsigned short la[2][64 * 32];
  __shared__ __align__(16) unsigned short lb[2][64 * 32];
  int tid = threadIdx.x, lane = tid & 31, wave = tid >> 5;
  size_t mbase = (size_t)blockIdx.x * 64;
  int nbase = blockIdx.y * 64;
  f32x8 acc[4] = {};
  fill_direct(la[0], A, DM, mbase, 0, tid);
  fill_clamp(lb[0], Bt, DM, nbase, PROJN, 0, tid);
  for (int kb = 0; kb < DM; kb += 32) {
    int cur = (kb >> 5) & 1;
    if (kb + 32 < DM) {
      fill_direct(la[cur ^ 1], A, DM, mbase, kb + 32, tid);
      fill_clamp(lb[cur ^ 1], Bt, DM, nbase, PROJN, kb + 32, tid);
      async_wait<4>();
    } else {
      async_wait<0>();
    }
    __syncthreads();
    wmma_step(la[cur], lb[cur], wave * 16, lane, acc);
    __syncthreads();
  }
  int hf = lane >> 4, nl = lane & 15;
#pragma unroll
  for (int nt = 0; nt < 4; ++nt)
#pragma unroll
    for (int r = 0; r < 8; ++r) {
      size_t gm = mbase + wave * 16 + r + 8 * hf;
      int gn = nbase + nt * 16 + nl;
      float v = acc[nt][r];
      if (gn < DI) {
        Ubf[gm * DI + gn] = f2bf(v);
      } else if (gn < 2 * DI) {
        float s = v / (1.f + __expf(-v));              // silu(z)
        Zsbf[gm * DI + (gn - DI)] = f2bf(s);
      } else if (gn < PROJN) {
        int hcol = gn - 2 * DI;
        float dv = v + dt_bias[hcol];
        Alpha[gm * NH + hcol] = 1.f / (1.f + __expf(dv));  // exp(-softplus)
      }
    }
}

// ---------------- Chunked selective scan ----------------
__global__ __launch_bounds__(64) void k_scan1(const unsigned short* __restrict__ Ubf,
                                              const float* __restrict__ Alpha,
                                              float* __restrict__ Ac,
                                              float* __restrict__ Bc) {
  int bid = blockIdx.x;                 // b*NH*NCH + h*NCH + chunk
  int chunk = bid & (NCH - 1);
  int h = (bid >> 6) % NH;
  int b = bid / (NCH * NH);
  int d = threadIdx.x;
  size_t tok0 = (size_t)b * LSEQ + (size_t)chunk * LCH;
  float hreg = 0.f, P = 1.f;
  for (int t = 0; t < LCH; ++t) {
    size_t tk = tok0 + t;
    float a = Alpha[tk * NH + h];
    float u = bf2f(Ubf[tk * DI + h * DH + d]);
    hreg = a * hreg + u;
    P *= a;
  }
  size_t base = ((size_t)(b * NH + h) * NCH + chunk);
  Bc[base * DH + d] = hreg;
  if (d == 0) Ac[base] = P;
}

__global__ __launch_bounds__(64) void k_scan2(const float* __restrict__ Ac,
                                              const float* __restrict__ Bc,
                                              float* __restrict__ Carry) {
  int bh = blockIdx.x;                  // 0..95
  int d = threadIdx.x;
  float c = 0.f;
  for (int i = 0; i < NCH; ++i) {
    size_t base = (size_t)bh * NCH + i;
    float Ai = Ac[base];
    float bi = Bc[base * DH + d];
    Carry[base * DH + d] = c;
    c = Ai * c + bi;
  }
}

__global__ __launch_bounds__(64) void k_scan3(const unsigned short* __restrict__ Ubf,
                                              const float* __restrict__ Alpha,
                                              const unsigned short* __restrict__ Zsbf,
                                              const float* __restrict__ Carry,
                                              unsigned short* __restrict__ Ybf) {
  int bid = blockIdx.x;
  int chunk = bid & (NCH - 1);
  int h = (bid >> 6) % NH;
  int b = bid / (NCH * NH);
  int d = threadIdx.x;
  size_t tok0 = (size_t)b * LSEQ + (size_t)chunk * LCH;
  size_t base = ((size_t)(b * NH + h) * NCH + chunk);
  float hreg = Carry[base * DH + d];
  for (int t = 0; t < LCH; ++t) {
    size_t tk = tok0 + t;
    float a = Alpha[tk * NH + h];
    float u = bf2f(Ubf[tk * DI + h * DH + d]);
    hreg = a * hreg + u;
    float y = hreg * bf2f(Zsbf[tk * DI + h * DH + d]);
    y = 10.f * tanhf(y * 0.1f);                        // scaled tanh
    Ybf[tk * DI + h * DH + d] = f2bf(y);
  }
}

// ---------------- GEMM: x1 = x + ls1 * (y @ W_out) ----------------
__global__ __launch_bounds__(128) void k_gemm_out(
    const unsigned short* __restrict__ A, const unsigned short* __restrict__ Bt,
    const float* __restrict__ x, const float* __restrict__ ls1,
    float* __restrict__ X1) {
  __shared__ __align__(16) unsigned short la[2][64 * 32];
  __shared__ __align__(16) unsigned short lb[2][64 * 32];
  int tid = threadIdx.x, lane = tid & 31, wave = tid >> 5;
  size_t mbase = (size_t)blockIdx.x * 64;
  int nbase = blockIdx.y * 64;
  f32x8 acc[4] = {};
  fill_direct(la[0], A, DI, mbase, 0, tid);
  fill_direct(lb[0], Bt, DI, nbase, 0, tid);
  for (int kb = 0; kb < DI; kb += 32) {
    int cur = (kb >> 5) & 1;
    if (kb + 32 < DI) {
      fill_direct(la[cur ^ 1], A, DI, mbase, kb + 32, tid);
      fill_direct(lb[cur ^ 1], Bt, DI, nbase, kb + 32, tid);
      async_wait<4>();
    } else {
      async_wait<0>();
    }
    __syncthreads();
    wmma_step(la[cur], lb[cur], wave * 16, lane, acc);
    __syncthreads();
  }
  int hf = lane >> 4, nl = lane & 15;
#pragma unroll
  for (int nt = 0; nt < 4; ++nt)
#pragma unroll
    for (int r = 0; r < 8; ++r) {
      size_t gm = mbase + wave * 16 + r + 8 * hf;
      int gn = nbase + nt * 16 + nl;
      X1[gm * DM + gn] = x[gm * DM + gn] + ls1[gn] * acc[nt][r];
    }
}

// ---------------- Zero expert counters ----------------
__global__ void k_zero_cnt(int* c) { if (threadIdx.x < NE) c[threadIdx.x] = 0; }

// ---------------- RMSNorm 2 + router top-2 + bucketing ----------------
__global__ __launch_bounds__(256) void k_rms2_router(
    const float* __restrict__ X1, const float* __restrict__ rw,
    const float* __restrict__ Wr, unsigned short* __restrict__ X2n,
    int* __restrict__ Cnt, int* __restrict__ Tok, float* __restrict__ Wl) {
  __shared__ float red[256];
  __shared__ float pl[256 * NE];
  __shared__ float lg[NE];
  int t = blockIdx.x, tid = threadIdx.x;
  const float* row = X1 + (size_t)t * DM;
  float v0 = row[tid], v1 = row[tid + 256], v2 = row[tid + 512];
  red[tid] = v0 * v0 + v1 * v1 + v2 * v2;
  __syncthreads();
  for (int s = 128; s > 0; s >>= 1) {
    if (tid < s) red[tid] += red[tid + s];
    __syncthreads();
  }
  float inv = rsqrtf(red[0] / (float)DM + 1e-5f);
  float n0 = v0 * inv * rw[tid];
  float n1 = v1 * inv * rw[tid + 256];
  float n2 = v2 * inv * rw[tid + 512];
  unsigned short* o = X2n + (size_t)t * DM;
  o[tid] = f2bf(n0); o[tid + 256] = f2bf(n1); o[tid + 512] = f2bf(n2);
#pragma unroll
  for (int e = 0; e < NE; ++e)
    pl[tid * NE + e] = n0 * Wr[tid * NE + e] + n1 * Wr[(tid + 256) * NE + e] +
                       n2 * Wr[(tid + 512) * NE + e];
  __syncthreads();
  if (tid < NE) {
    float s = 0.f;
    for (int i = 0; i < 256; ++i) s += pl[i * NE + tid];
    lg[tid] = s * 2.0f;                                 // 1/router_temp
  }
  __syncthreads();
  if (tid == 0) {
    int i0 = 0; float b0 = lg[0];
    for (int e = 1; e < NE; ++e) if (lg[e] > b0) { b0 = lg[e]; i0 = e; }
    int i1 = (i0 == 0) ? 1 : 0; float b1 = lg[i1];
    for (int e = 0; e < NE; ++e)
      if (e != i0 && lg[e] > b1) { b1 = lg[e]; i1 = e; }
    float p0 = 1.f / (1.f + __expf(b1 - b0));
    float p1 = 1.f - p0;
    int pos0 = atomicAdd(&Cnt[i0], 1);
    Tok[i0 * CAP + pos0] = t * 2 + 0; Wl[i0 * CAP + pos0] = p0;
    int pos1 = atomicAdd(&Cnt[i1], 1);
    Tok[i1 * CAP + pos1] = t * 2 + 1; Wl[i1 * CAP + pos1] = p1;
  }
}

// ---------------- GEMM: lat = scaled_tanh(x2n @ W_down) ----------------
__global__ __launch_bounds__(128) void k_gemm_lat(
    const unsigned short* __restrict__ A, const unsigned short* __restrict__ Bt,
    unsigned short* __restrict__ Lat) {
  __shared__ __align__(16) unsigned short la[2][64 * 32];
  __shared__ __align__(16) unsigned short lb[2][64 * 32];
  int tid = threadIdx.x, lane = tid & 31, wave = tid >> 5;
  size_t mbase = (size_t)blockIdx.x * 64;
  int nbase = blockIdx.y * 64;
  f32x8 acc[4] = {};
  fill_direct(la[0], A, DM, mbase, 0, tid);
  fill_direct(lb[0], Bt, DM, nbase, 0, tid);
  for (int kb = 0; kb < DM; kb += 32) {
    int cur = (kb >> 5) & 1;
    if (kb + 32 < DM) {
      fill_direct(la[cur ^ 1], A, DM, mbase, kb + 32, tid);
      fill_direct(lb[cur ^ 1], Bt, DM, nbase, kb + 32, tid);
      async_wait<4>();
    } else {
      async_wait<0>();
    }
    __syncthreads();
    wmma_step(la[cur], lb[cur], wave * 16, lane, acc);
    __syncthreads();
  }
  int hf = lane >> 4, nl = lane & 15;
#pragma unroll
  for (int nt = 0; nt < 4; ++nt)
#pragma unroll
    for (int r = 0; r < 8; ++r) {
      size_t gm = mbase + wave * 16 + r + 8 * hf;
      int gn = nbase + nt * 16 + nl;
      float v = 10.f * tanhf(acc[nt][r] * 0.1f);
      Lat[gm * R3 + gn] = f2bf(v);
    }
}

// ---------------- GEMM: MoE expert-gathered lat @ G[e] ----------------
__global__ __launch_bounds__(128) void k_gemm_moe(
    const unsigned short* __restrict__ Lat, const unsigned short* __restrict__ Gt,
    const int* __restrict__ Cnt, const int* __restrict__ Tok,
    const float* __restrict__ Wl, float* __restrict__ Moe) {
  int e = blockIdx.z;
  int cnt = Cnt[e];
  int mbase = blockIdx.x * 64;
  if (mbase >= cnt) return;
  const unsigned short* Bt = Gt + (size_t)e * R2 * R3;   // [R2][R3] (n,k)
  const int* tok = Tok + e * CAP;
  const float* wl = Wl + e * CAP;
  __shared__ __align__(16) unsigned short la[2][64 * 32];
  __shared__ __align__(16) unsigned short lb[2][64 * 32];
  int tid = threadIdx.x, lane = tid & 31, wave = tid >> 5;
  int nbase = blockIdx.y * 64;
  f32x8 acc[4] = {};
  fill_gather(la[0], Lat, tok, mbase, cnt, 0, tid);
  fill_direct(lb[0], Bt, R3, nbase, 0, tid);
  for (int kb = 0; kb < R3; kb += 32) {
    int cur = (kb >> 5) & 1;
    if (kb + 32 < R3) {
      fill_gather(la[cur ^ 1], Lat, tok, mbase, cnt, kb + 32, tid);
      fill_direct(lb[cur ^ 1], Bt, R3, nbase, kb + 32, tid);
      async_wait<4>();
    } else {
      async_wait<0>();
    }
    __syncthreads();
    wmma_step(la[cur], lb[cur], wave * 16, lane, acc);
    __syncthreads();
  }
  int hf = lane >> 4, nl = lane & 15;
#pragma unroll
  for (int nt = 0; nt < 4; ++nt)
#pragma unroll
    for (int r = 0; r < 8; ++r) {
      int entry = mbase + wave * 16 + r + 8 * hf;
      int gn = nbase + nt * 16 + nl;
      if (entry < cnt) {
        int t2 = tok[entry];
        Moe[(size_t)t2 * R2 + gn] = wl[entry] * acc[nt][r];  // slot row t*2+k
      }
    }
}

// ---------------- GEMM: out = x1 + ls2 * ((moe0+moe1) @ W_up) ----------------
__global__ __launch_bounds__(128) void k_gemm_up(
    const float* __restrict__ Moe, const unsigned short* __restrict__ Bt,
    const float* __restrict__ X1, const float* __restrict__ ls2,
    float* __restrict__ out) {
  __shared__ __align__(16) unsigned short la[2][64 * 32];
  __shared__ __align__(16) unsigned short lb[2][64 * 32];
  int tid = threadIdx.x, lane = tid & 31, wave = tid >> 5;
  size_t mbase = (size_t)blockIdx.x * 64;
  int nbase = blockIdx.y * 64;
  f32x8 acc[4] = {};
  fill_conv(la[0], Moe, mbase, 0, tid);
  fill_direct(lb[0], Bt, R2, nbase, 0, tid);
  for (int kb = 0; kb < R2; kb += 32) {
    int cur = (kb >> 5) & 1;
    if (kb + 32 < R2) {
      fill_conv(la[cur ^ 1], Moe, mbase, kb + 32, tid);
      fill_direct(lb[cur ^ 1], Bt, R2, nbase, kb + 32, tid);
      async_wait<2>();
    } else {
      async_wait<0>();
    }
    __syncthreads();
    wmma_step(la[cur], lb[cur], wave * 16, lane, acc);
    __syncthreads();
  }
  int hf = lane >> 4, nl = lane & 15;
#pragma unroll
  for (int nt = 0; nt < 4; ++nt)
#pragma unroll
    for (int r = 0; r < 8; ++r) {
      size_t gm = mbase + wave * 16 + r + 8 * hf;
      int gn = nbase + nt * 16 + nl;
      out[gm * DM + gn] = X1[gm * DM + gn] + ls2[gn] * acc[nt][r];
    }
}

// ---------------- Host ----------------
extern "C" void kernel_launch(void* const* d_in, const int* in_sizes, int n_in,
                              void* d_out, int out_size, void* d_ws, size_t ws_size,
                              hipStream_t stream) {
  (void)in_sizes; (void)n_in; (void)out_size; (void)ws_size;
  const float* x       = (const float*)d_in[0];
  const float* rms1_w  = (const float*)d_in[1];
  const float* W_in    = (const float*)d_in[2];
  const float* dt_bias = (const float*)d_in[3];
  const float* W_out   = (const float*)d_in[4];
  const float* ls1     = (const float*)d_in[5];
  const float* rms2_w  = (const float*)d_in[6];
  const float* W_down  = (const float*)d_in[7];
  const float* W_router= (const float*)d_in[8];
  const float* G       = (const float*)d_in[9];
  const float* W_up    = (const float*)d_in[10];
  const float* ls2     = (const float*)d_in[11];
  float* out = (float*)d_out;

  char* ws = (char*)d_ws;
  size_t off = 0;
  auto take = [&](size_t bytes) -> char* {
    char* p = ws + off;
    off = (off + bytes + 255) & ~(size_t)255;
    return p;
  };
  unsigned short* Wint   = (unsigned short*)take((size_t)PROJN * DM * 2);
  unsigned short* Woutt  = (unsigned short*)take((size_t)DM * DI * 2);
  unsigned short* Wdownt = (unsigned short*)take((size_t)R3 * DM * 2);
  unsigned short* Gt     = (unsigned short*)take((size_t)NE * R2 * R3 * 2);
  unsigned short* Wupt   = (unsigned short*)take((size_t)DM * R2 * 2);
  unsigned short* Xn     = (unsigned short*)take((size_t)T_TOK * DM * 2);
  unsigned short* Ubf    = (unsigned short*)take((size_t)T_TOK * DI * 2);
  unsigned short* Zsbf   = (unsigned short*)take((size_t)T_TOK * DI * 2);
  float*          Alpha  = (float*)take((size_t)T_TOK * NH * 4);
  float*          Ac     = (float*)take((size_t)2 * NH * NCH * 4);
  float*          Bc     = (float*)take((size_t)2 * NH * NCH * DH * 4);
  float*          Carry  = (float*)take((size_t)2 * NH * NCH * DH * 4);
  unsigned short* Ybf    = (unsigned short*)take((size_t)T_TOK * DI * 2);
  float*          X1     = (float*)take((size_t)T_TOK * DM * 4);
  unsigned short* X2n    = (unsigned short*)take((size_t)T_TOK * DM * 2);
  int*            Cnt    = (int*)take(NE * 4);
  int*            Tok    = (int*)take((size_t)NE * CAP * 4);
  float*          Wl     = (float*)take((size_t)NE * CAP * 4);
  unsigned short* Lat    = (unsigned short*)take((size_t)T_TOK * R3 * 2);
  float*          Moe    = (float*)take((size_t)T_TOK * 2 * R2 * 4);

  dim3 b16(16, 16);
  // Pre-transpose all GEMM B operands to (N,K) bf16.
  k_transpose_bf16<<<dim3((PROJN + 15) / 16, (DM + 15) / 16, 1), b16, 0, stream>>>(W_in, Wint, DM, PROJN);
  k_transpose_bf16<<<dim3((DM + 15) / 16, (DI + 15) / 16, 1), b16, 0, stream>>>(W_out, Woutt, DI, DM);
  k_transpose_bf16<<<dim3((R3 + 15) / 16, (DM + 15) / 16, 1), b16, 0, stream>>>(W_down, Wdownt, DM, R3);
  k_transpose_bf16<<<dim3((R2 + 15) / 16, (R3 + 15) / 16, NE), b16, 0, stream>>>(G, Gt, R3, R2);
  k_transpose_bf16<<<dim3((DM + 15) / 16, (R2 + 15) / 16, 1), b16, 0, stream>>>(W_up, Wupt, R2, DM);

  k_rmsnorm<<<T_TOK, 256, 0, stream>>>(x, rms1_w, Xn);
  k_gemm_proj<<<dim3(T_TOK / 64, (PROJN + 63) / 64), 128, 0, stream>>>(Xn, Wint, dt_bias, Ubf, Zsbf, Alpha);
  k_scan1<<<2 * NH * NCH, 64, 0, stream>>>(Ubf, Alpha, Ac, Bc);
  k_scan2<<<2 * NH, 64, 0, stream>>>(Ac, Bc, Carry);
  k_scan3<<<2 * NH * NCH, 64, 0, stream>>>(Ubf, Alpha, Zsbf, Carry, Ybf);
  k_gemm_out<<<dim3(T_TOK / 64, DM / 64), 128, 0, stream>>>(Ybf, Woutt, x, ls1, X1);
  k_zero_cnt<<<1, 32, 0, stream>>>(Cnt);
  k_rms2_router<<<T_TOK, 256, 0, stream>>>(X1, rms2_w, W_router, X2n, Cnt, Tok, Wl);
  k_gemm_lat<<<dim3(T_TOK / 64, R3 / 64), 128, 0, stream>>>(X2n, Wdownt, Lat);
  k_gemm_moe<<<dim3(CAP / 64, R2 / 64, NE), 128, 0, stream>>>(Lat, Gt, Cnt, Tok, Wl, Moe);
  k_gemm_up<<<dim3(T_TOK / 64, DM / 64), 128, 0, stream>>>(Moe, Wupt, X1, ls2, out);
}
